// ProcessorMessagePassing_39298950758846
// MI455X (gfx1250) — matrix operand order repaired and checked by reading, not compiled, not run on hardware
//
#include <hip/hip_runtime.h>
#include <hip/hip_bf16.h>

typedef __attribute__((ext_vector_type(16))) _Float16 v16h;
typedef __attribute__((ext_vector_type(8)))  float    v8f;
typedef __attribute__((ext_vector_type(2)))  __fp16   fp16x2;

#define HDIM 128
#define LN_EPS 1e-5f

// LDS strides (in dwords) chosen for ds_load_b128 alignment (multiple of 4)
// and bank spread (stride mod 64 cycles through distinct banks per row/col).
#define SA_STRIDE 20   // A chunk: 16 pair-dwords + 4 pad
#define SB_STRIDE 20   // B chunk (fragment-major): 16 pair-dwords + 4 pad
#define SH_STRIDE 68   // hidden: 64 pair-dwords + 4 pad  (68*row mod 64 = 4*row)

union F16x16 { uint4 q[2]; v16h v; };

__device__ __forceinline__ unsigned pk_f16(float a, float b) {
#if __has_builtin(__builtin_amdgcn_cvt_pkrtz)
    union { fp16x2 h; unsigned u; } t;
    t.h = __builtin_amdgcn_cvt_pkrtz(a, b);
    return t.u;
#else
    union { _Float16 h[2]; unsigned u; } t;
    t.h[0] = (_Float16)a; t.h[1] = (_Float16)b;
    return t.u;
#endif
}

// One 16(M)x128(N)x32(K) WMMA step for a wave.
// A: row-major pair-dwords in LDS, rowStride dwords per row (pairs {0..3}+4h, {8..11}+4h).
// B: fragment-major chunk: sBq[col*SB_STRIDE + 4*half + {0..3, 8..11}] -> two 4-dword runs, same pairs as A.
__device__ __forceinline__ void mma_chunk(const unsigned* sAq, int rowStride,
                                          int rowBase, int pairBase,
                                          const unsigned* sBq,
                                          v8f* acc, int L, int half) {
    F16x16 a;
    const unsigned* ar = sAq + (rowBase + L) * rowStride + pairBase + 4 * half;
    a.q[0] = *(const uint4*)(ar);
    a.q[1] = *(const uint4*)(ar + 8);
#pragma unroll
    for (int t = 0; t < 8; ++t) {
        F16x16 b;
        const unsigned* br = sBq + (t * 16 + L) * SB_STRIDE + 4 * half;
        b.q[0] = *(const uint4*)(br);
        b.q[1] = *(const uint4*)(br + 8);
        acc[t] = __builtin_amdgcn_wmma_f32_16x16x32_f16(
            false, a.v, false, b.v, (short)0, acc[t], false, false);
    }
}

// Pack fp32 row-major [K][128] weights to f16 pair-dwords: wp[(k/2)*128+col] = (w[k][col], w[k+1][col])
__global__ void pack_w_kernel(const float* __restrict__ w, unsigned* __restrict__ wp, int Kp) {
    int i = blockIdx.x * 256 + threadIdx.x;
    if (i < Kp * 128) {
        int col = i & 127, kp = i >> 7;
        wp[i] = pk_f16(w[(2 * kp) * 128 + col], w[(2 * kp + 1) * 128 + col]);
    }
}

// Stage one 32-K chunk of packed weights into fragment-major LDS layout.
__device__ __forceinline__ void stage_B(unsigned* sB, const unsigned* wsrc, int tid) {
#pragma unroll
    for (int p = 0; p < 8; ++p) {
        const int idx = tid + p * 256;      // 0..2047
        const int col = idx & 127;
        const int pp  = idx >> 7;           // 0..15 local pair
        sB[col * SB_STRIDE + pp] = wsrc[pp * 128 + col];
    }
}

// ---------------- Edge block: e_new = LN(silu(cat(x[src],x[dst],ea)@W1+b1)@W2+b2)*g+beta
// out_e = ea + e_new ; atomicAdd agg[dst] += e_new
__global__ __launch_bounds__(256) void edge_kernel(
    const float* __restrict__ x, const float* __restrict__ edge_attr,
    const int* __restrict__ eidx, int E,
    const unsigned* __restrict__ w1p, const unsigned* __restrict__ w2p,
    const float* __restrict__ b1, const float* __restrict__ b2,
    const float* __restrict__ g, const float* __restrict__ bet,
    float* __restrict__ out_e, float* __restrict__ agg)
{
    __shared__ __align__(16) unsigned sA[128 * SA_STRIDE];
    __shared__ __align__(16) unsigned sB[128 * SB_STRIDE];
    __shared__ __align__(16) unsigned sH[128 * SH_STRIDE];
    __shared__ unsigned sOff[3 * 128];   // 32-bit byte offsets per row per group
    __shared__ int sDstI[128];

    const int tid = threadIdx.x;
    const int lane = tid & 31, wave = tid >> 5;
    const int L = lane & 15, half = lane >> 4;
    const long long e0 = (long long)blockIdx.x * 128;
    const int* srcI = eidx;
    const int* dstI = eidx + E;

    // Resolve all gather offsets for this 128-edge tile once (byte offsets fit in 32 bits).
    if (tid < 128) {
        long long e = e0 + tid; if (e >= E) e = E - 1;
        const int s = srcI[e], d = dstI[e];
        sOff[tid]       = (unsigned)s * (HDIM * 4u);
        sOff[128 + tid] = (unsigned)d * (HDIM * 4u);
        sOff[256 + tid] = (unsigned)e * (HDIM * 4u);
        sDstI[tid] = d;
    }

    v8f acc[8];
#pragma unroll
    for (int t = 0; t < 8; ++t) { v8f z = {0,0,0,0,0,0,0,0}; acc[t] = z; }

    // GEMM1: K = 384 (12 chunks of 32). cols [0,128)=x[src], [128,256)=x[dst], [256,384)=edge_attr
    for (int c = 0; c < 12; ++c) {
        __syncthreads();                        // also makes sOff visible (c==0)
        const int grp = c >> 2, col0 = (c & 3) * 32;
        const char* base = (grp == 2) ? (const char*)edge_attr : (const char*)x;
#pragma unroll
        for (int p = 0; p < 4; ++p) {
            const int idx = tid + p * 256;      // 0..1023
            const int row = idx >> 3;           // 0..127
            const int f4 = idx & 7;             // 0..7 float4's
            const float* rp = (const float*)(base + sOff[grp * 128 + row]);
            const float4 vv = *(const float4*)(rp + col0 + f4 * 4);
            sA[row * SA_STRIDE + f4 * 2]     = pk_f16(vv.x, vv.y);
            sA[row * SA_STRIDE + f4 * 2 + 1] = pk_f16(vv.z, vv.w);
        }
        stage_B(sB, w1p + c * 16 * 128, tid);
        __syncthreads();
        mma_chunk(sA, SA_STRIDE, wave * 16, 0, sB, acc, L, half);
    }

    // bias1 + SiLU -> sH (f16), pair-dword layout with padded row stride
    float b1v[8];
#pragma unroll
    for (int t = 0; t < 8; ++t) b1v[t] = b1[t * 16 + L];
    _Float16* sH16 = (_Float16*)sH;
#pragma unroll
    for (int t = 0; t < 8; ++t) {
#pragma unroll
        for (int v = 0; v < 8; ++v) {
            float h = acc[t][v] + b1v[t];
            h = h / (1.f + __expf(-h));
            const int row = wave * 16 + v + 8 * half;
            const int col = t * 16 + L;
            sH16[row * (2 * SH_STRIDE) + col] = (_Float16)h;
        }
    }

    // GEMM2: K = 128 (4 chunks)
    v8f acc2[8];
#pragma unroll
    for (int t = 0; t < 8; ++t) { v8f z = {0,0,0,0,0,0,0,0}; acc2[t] = z; }
    for (int c = 0; c < 4; ++c) {
        __syncthreads();                       // guards sH writes (c==0) and sB WAR
        stage_B(sB, w2p + c * 16 * 128, tid);
        __syncthreads();
        mma_chunk(sH, SH_STRIDE, wave * 16, c * 16, sB, acc2, L, half);
    }

    // bias2 + LayerNorm (per-row stats via shfl_xor over the 16-lane half)
    float gv[8], bev[8];
#pragma unroll
    for (int t = 0; t < 8; ++t) {
        const float b2v = b2[t * 16 + L];
        gv[t] = g[t * 16 + L]; bev[t] = bet[t * 16 + L];
#pragma unroll
        for (int v = 0; v < 8; ++v) acc2[t][v] += b2v;
    }
    float mean[8], rstd[8];
#pragma unroll
    for (int v = 0; v < 8; ++v) {
        float s = 0.f, q = 0.f;
#pragma unroll
        for (int t = 0; t < 8; ++t) { const float xv = acc2[t][v]; s += xv; q += xv * xv; }
#pragma unroll
        for (int m = 1; m <= 8; m <<= 1) { s += __shfl_xor(s, m); q += __shfl_xor(q, m); }
        const float mu = s * (1.f / 128.f);
        const float var = q * (1.f / 128.f) - mu * mu;
        mean[v] = mu; rstd[v] = rsqrtf(var + LN_EPS);
    }

    // epilogue: residual store + scatter-add
#pragma unroll
    for (int v = 0; v < 8; ++v) {
        const int row = wave * 16 + v + 8 * half;
        const long long e = e0 + row;
        if (e < E) {
            const long long d = (long long)sDstI[row] * HDIM;
            const float* ea = edge_attr + e * HDIM;
            float* oe = out_e + e * HDIM;
#pragma unroll
            for (int t = 0; t < 8; ++t) {
                const int col = t * 16 + L;
                const float valn = (acc2[t][v] - mean[v]) * rstd[v] * gv[t] + bev[t];
                oe[col] = ea[col] + valn;
                atomicAdd(agg + d + col, valn);
            }
        }
    }
}

// ---------------- Node block: x_new = LN(silu(cat(x,agg)@W1+b1)@W2+b2)*g+beta ; out_n = x + x_new
__global__ __launch_bounds__(256) void node_kernel(
    const float* __restrict__ x, const float* __restrict__ agg, int N,
    const unsigned* __restrict__ w1p, const unsigned* __restrict__ w2p,
    const float* __restrict__ b1, const float* __restrict__ b2,
    const float* __restrict__ g, const float* __restrict__ bet,
    float* __restrict__ out_n)
{
    __shared__ __align__(16) unsigned sA[128 * SA_STRIDE];
    __shared__ __align__(16) unsigned sB[128 * SB_STRIDE];
    __shared__ __align__(16) unsigned sH[128 * SH_STRIDE];

    const int tid = threadIdx.x;
    const int lane = tid & 31, wave = tid >> 5;
    const int L = lane & 15, half = lane >> 4;
    const long long n0 = (long long)blockIdx.x * 128;

    v8f acc[8];
#pragma unroll
    for (int t = 0; t < 8; ++t) { v8f z = {0,0,0,0,0,0,0,0}; acc[t] = z; }

    // GEMM1: K = 256 (8 chunks). cols [0,128)=x, [128,256)=agg
    for (int c = 0; c < 8; ++c) {
        __syncthreads();
        const int grp = c >> 2, col0 = (c & 3) * 32;
#pragma unroll
        for (int p = 0; p < 4; ++p) {
            const int idx = tid + p * 256;
            const int row = idx >> 3;
            const int f4 = idx & 7;
            long long n = n0 + row; if (n >= N) n = N - 1;
            const float* rp = (grp == 0 ? x : agg) + n * HDIM;
            const float4 vv = *(const float4*)(rp + col0 + f4 * 4);
            sA[row * SA_STRIDE + f4 * 2]     = pk_f16(vv.x, vv.y);
            sA[row * SA_STRIDE + f4 * 2 + 1] = pk_f16(vv.z, vv.w);
        }
        stage_B(sB, w1p + c * 16 * 128, tid);
        __syncthreads();
        mma_chunk(sA, SA_STRIDE, wave * 16, 0, sB, acc, L, half);
    }

    float b1v[8];
#pragma unroll
    for (int t = 0; t < 8; ++t) b1v[t] = b1[t * 16 + L];
    _Float16* sH16 = (_Float16*)sH;
#pragma unroll
    for (int t = 0; t < 8; ++t) {
#pragma unroll
        for (int v = 0; v < 8; ++v) {
            float h = acc[t][v] + b1v[t];
            h = h / (1.f + __expf(-h));
            const int row = wave * 16 + v + 8 * half;
            const int col = t * 16 + L;
            sH16[row * (2 * SH_STRIDE) + col] = (_Float16)h;
        }
    }

    v8f acc2[8];
#pragma unroll
    for (int t = 0; t < 8; ++t) { v8f z = {0,0,0,0,0,0,0,0}; acc2[t] = z; }
    for (int c = 0; c < 4; ++c) {
        __syncthreads();
        stage_B(sB, w2p + c * 16 * 128, tid);
        __syncthreads();
        mma_chunk(sH, SH_STRIDE, wave * 16, c * 16, sB, acc2, L, half);
    }

    float gv[8], bev[8];
#pragma unroll
    for (int t = 0; t < 8; ++t) {
        const float b2v = b2[t * 16 + L];
        gv[t] = g[t * 16 + L]; bev[t] = bet[t * 16 + L];
#pragma unroll
        for (int v = 0; v < 8; ++v) acc2[t][v] += b2v;
    }
    float mean[8], rstd[8];
#pragma unroll
    for (int v = 0; v < 8; ++v) {
        float s = 0.f, q = 0.f;
#pragma unroll
        for (int t = 0; t < 8; ++t) { const float xv = acc2[t][v]; s += xv; q += xv * xv; }
#pragma unroll
        for (int m = 1; m <= 8; m <<= 1) { s += __shfl_xor(s, m); q += __shfl_xor(q, m); }
        const float mu = s * (1.f / 128.f);
        const float var = q * (1.f / 128.f) - mu * mu;
        mean[v] = mu; rstd[v] = rsqrtf(var + LN_EPS);
    }

#pragma unroll
    for (int v = 0; v < 8; ++v) {
        const int row = wave * 16 + v + 8 * half;
        const long long n = n0 + row;
        if (n < N) {
            const float* xr = x + n * HDIM;
            float* on = out_n + n * HDIM;
#pragma unroll
            for (int t = 0; t < 8; ++t) {
                const int col = t * 16 + L;
                const float valn = (acc2[t][v] - mean[v]) * rstd[v] * gv[t] + bev[t];
                on[col] = xr[col] + valn;
            }
        }
    }
}

extern "C" void kernel_launch(void* const* d_in, const int* in_sizes, int n_in,
                              void* d_out, int out_size, void* d_ws, size_t ws_size,
                              hipStream_t stream) {
    const float* x         = (const float*)d_in[0];
    const float* edge_attr = (const float*)d_in[1];
    const int*   eidx      = (const int*)d_in[2];
    const float* eb_w1 = (const float*)d_in[3];
    const float* eb_b1 = (const float*)d_in[4];
    const float* eb_w2 = (const float*)d_in[5];
    const float* eb_b2 = (const float*)d_in[6];
    const float* eb_g  = (const float*)d_in[7];
    const float* eb_be = (const float*)d_in[8];
    const float* nb_w1 = (const float*)d_in[9];
    const float* nb_b1 = (const float*)d_in[10];
    const float* nb_w2 = (const float*)d_in[11];
    const float* nb_b2 = (const float*)d_in[12];
    const float* nb_g  = (const float*)d_in[13];
    const float* nb_be = (const float*)d_in[14];

    const int N = in_sizes[0] / HDIM;
    const int E = in_sizes[1] / HDIM;

    float* out_n = (float*)d_out;
    float* out_e = (float*)d_out + (size_t)N * HDIM;

    // workspace: agg[N*H] f32, then packed f16 weights (pair-dwords)
    char* ws = (char*)d_ws;
    float* agg = (float*)ws;
    const size_t aggB = (size_t)N * HDIM * sizeof(float);
    unsigned* w1e = (unsigned*)(ws + aggB);   // 192*128 dwords (K=384)
    unsigned* w2e = w1e + 192 * 128;          //  64*128 dwords (K=128)
    unsigned* w1n = w2e + 64 * 128;           // 128*128 dwords (K=256)
    unsigned* w2n = w1n + 128 * 128;          //  64*128 dwords (K=128)

    (void)hipMemsetAsync(agg, 0, aggB, stream);

    pack_w_kernel<<<(192 * 128 + 255) / 256, 256, 0, stream>>>(eb_w1, w1e, 192);
    pack_w_kernel<<<(64  * 128 + 255) / 256, 256, 0, stream>>>(eb_w2, w2e, 64);
    pack_w_kernel<<<(128 * 128 + 255) / 256, 256, 0, stream>>>(nb_w1, w1n, 128);
    pack_w_kernel<<<(64  * 128 + 255) / 256, 256, 0, stream>>>(nb_w2, w2n, 64);

    edge_kernel<<<(E + 127) / 128, 256, 0, stream>>>(
        x, edge_attr, eidx, E, w1e, w2e, eb_b1, eb_b2, eb_g, eb_be, out_e, agg);

    node_kernel<<<(N + 127) / 128, 256, 0, stream>>>(
        x, agg, N, w1n, w2n, nb_b1, nb_b2, nb_g, nb_be, out_n);
}
